// MultiScaleAttention_2877628088801
// MI455X (gfx1250) — matrix operand order repaired and checked by reading, not compiled, not run on hardware
//
#include <hip/hip_runtime.h>
#include <hip/hip_bf16.h>

// ---------------- problem constants ----------------
#define BB    4
#define NH    8
#define BH    32          // BB*NH
#define DIMC  768
#define DH    96
#define T0c   8
#define H0c   28
#define W0c   28
#define NTOK  6273        // T0*H0*W0 + 1
#define To_   8
#define Ho_   14
#define Wo_   14
#define NP    1569        // To*Ho*Wo + 1
#define NPAD  1600        // padded key count for v-transposed buffer
#define MQKV  25092       // BB*NTOK
#define MPROJ 6276        // BB*NP

typedef __attribute__((ext_vector_type(16))) __bf16 bf16x16;
typedef __attribute__((ext_vector_type(8)))  float  f32x8;

// ---------------- bf16 bit helpers ----------------
static __device__ __forceinline__ float bf2f(unsigned short u) {
    unsigned int w = ((unsigned int)u) << 16;
    return __builtin_bit_cast(float, w);
}
static __device__ __forceinline__ unsigned short f2bf(float f) {
    unsigned int u = __builtin_bit_cast(unsigned int, f);
    unsigned int r = u + 0x7FFFu + ((u >> 16) & 1u);
    return (unsigned short)(r >> 16);
}
static __device__ __forceinline__ __bf16 us2bf(unsigned short u) {
    return __builtin_bit_cast(__bf16, u);
}

static __device__ __forceinline__ f32x8 wmma_bf16(bf16x16 a, bf16x16 b, f32x8 c) {
    return __builtin_amdgcn_wmma_f32_16x16x32_bf16(false, a, false, b, (short)0, c,
                                                   false, false);
}

// A-fragment (16x32, M x K): lane<16 -> row, K {k0..k0+15};
// lane>=16 -> same row set, K {k0+16..k0+31}. rowp points at lane's row.
static __device__ __forceinline__ bf16x16 load_a_frag(const unsigned short* rowp,
                                                      int k0, int hi) {
    bf16x16 v;
    int kb = k0 + hi * 16;
#pragma unroll
    for (int i = 0; i < 8; ++i) v[i]     = us2bf(rowp[kb + i]);
#pragma unroll
    for (int i = 0; i < 8; ++i) v[8 + i] = us2bf(rowp[kb + 8 + i]);
    return v;
}
// B-fragment: lane holds one N-column, 16 contiguous K values starting at p.
static __device__ __forceinline__ bf16x16 load_b_frag(const unsigned short* p) {
    bf16x16 v;
#pragma unroll
    for (int i = 0; i < 16; ++i) v[i] = us2bf(p[i]);
    return v;
}

// ---------------- utility kernels (4-wide vectorized) ----------------
__global__ void cvt_f32_bf16_v4(const float* __restrict__ s,
                                unsigned short* __restrict__ d, long n4) {
    long i = (long)blockIdx.x * blockDim.x + threadIdx.x;
    if (i < n4) {
        float4 f = ((const float4*)s)[i];
        ushort4 u;
        u.x = f2bf(f.x); u.y = f2bf(f.y); u.z = f2bf(f.z); u.w = f2bf(f.w);
        ((ushort4*)d)[i] = u;
    }
}
__global__ void zero_u16_v4(unsigned short* __restrict__ p, long n4) {
    long i = (long)blockIdx.x * blockDim.x + threadIdx.x;
    if (i < n4) ((ushort4*)p)[i] = make_ushort4(0, 0, 0, 0);
}

// ---------------- QKV projection GEMM: Y = X * W^T, split-head scatter ------
// 32x64 tile per wave: 2 A-frags, 8 accumulators, B-frags reused twice.
// X: (MQKV, 768) bf16 ; W: (768, 768) bf16 (o, d) ; Out: (BH, NTOK, 96)
__global__ void __launch_bounds__(256)
gemm_qkv(const unsigned short* __restrict__ Xb,
         const unsigned short* __restrict__ Wb,
         unsigned short* __restrict__ Outb, int Mrows) {
    int w = threadIdx.x >> 5, lane = threadIdx.x & 31;
    int hi = lane >> 4, l4 = lane & 15;
    int Mt = (Mrows + 31) >> 5;
    int mt = blockIdx.x * 8 + w;
    if (mt >= Mt) return;
    int m0 = mt * 32, n0 = blockIdx.y * 64;
    int r0 = m0 + l4;      if (r0 >= Mrows) r0 = Mrows - 1;
    int r1 = m0 + 16 + l4; if (r1 >= Mrows) r1 = Mrows - 1;
    const unsigned short* A0 = Xb + (long)r0 * DIMC;
    const unsigned short* A1 = Xb + (long)r1 * DIMC;
    f32x8 acc[2][4] = {};
    for (int k0 = 0; k0 < DIMC; k0 += 32) {
        __builtin_prefetch(A0 + k0 + 128, 0, 0);
        __builtin_prefetch(A1 + k0 + 128, 0, 0);
        bf16x16 a0 = load_a_frag(A0, k0, hi);
        bf16x16 a1 = load_a_frag(A1, k0, hi);
#pragma unroll
        for (int j = 0; j < 4; ++j) {
            bf16x16 bfr =
                load_b_frag(Wb + (long)(n0 + j * 16 + l4) * DIMC + k0 + hi * 16);
            acc[0][j] = wmma_bf16(a0, bfr, acc[0][j]);
            acc[1][j] = wmma_bf16(a1, bfr, acc[1][j]);
        }
    }
    // hoisted column decomposition (per j, loop-invariant over rows)
    int hcol[4], dcol[4];
#pragma unroll
    for (int j = 0; j < 4; ++j) {
        int o = n0 + j * 16 + l4;
        hcol[j] = o / DH;
        dcol[j] = o - hcol[j] * DH;
    }
    bool full = (m0 + 31) < Mrows;  // wave-uniform fast path
#pragma unroll
    for (int mh = 0; mh < 2; ++mh)
#pragma unroll
        for (int i = 0; i < 8; ++i) {
            int row = m0 + mh * 16 + i + 8 * hi;
            if (full || row < Mrows) {
                int bb = row / NTOK, n = row - bb * NTOK;
                long rbase = (long)bb * NH * NTOK + n;
#pragma unroll
                for (int j = 0; j < 4; ++j)
                    Outb[((long)hcol[j] * NTOK + rbase) * DH + dcol[j]] =
                        f2bf(acc[mh][j][i]);
            }
        }
}

// ---------------- output projection GEMM: out = A * Wproj^T + bias (f32) ---
__global__ void __launch_bounds__(256)
gemm_proj(const unsigned short* __restrict__ Ab,
          const unsigned short* __restrict__ Wb,
          const float* __restrict__ bias, float* __restrict__ Outf, int Mrows) {
    int w = threadIdx.x >> 5, lane = threadIdx.x & 31;
    int hi = lane >> 4, l4 = lane & 15;
    int Mt = (Mrows + 31) >> 5;
    int mt = blockIdx.x * 8 + w;
    if (mt >= Mt) return;
    int m0 = mt * 32, n0 = blockIdx.y * 64;
    int r0 = m0 + l4;      if (r0 >= Mrows) r0 = Mrows - 1;
    int r1 = m0 + 16 + l4; if (r1 >= Mrows) r1 = Mrows - 1;
    const unsigned short* A0 = Ab + (long)r0 * DIMC;
    const unsigned short* A1 = Ab + (long)r1 * DIMC;
    f32x8 acc[2][4] = {};
    for (int k0 = 0; k0 < DIMC; k0 += 32) {
        __builtin_prefetch(A0 + k0 + 128, 0, 0);
        __builtin_prefetch(A1 + k0 + 128, 0, 0);
        bf16x16 a0 = load_a_frag(A0, k0, hi);
        bf16x16 a1 = load_a_frag(A1, k0, hi);
#pragma unroll
        for (int j = 0; j < 4; ++j) {
            bf16x16 bfr =
                load_b_frag(Wb + (long)(n0 + j * 16 + l4) * DIMC + k0 + hi * 16);
            acc[0][j] = wmma_bf16(a0, bfr, acc[0][j]);
            acc[1][j] = wmma_bf16(a1, bfr, acc[1][j]);
        }
    }
    float bj[4];
#pragma unroll
    for (int j = 0; j < 4; ++j) bj[j] = bias[n0 + j * 16 + l4];
    bool full = (m0 + 31) < Mrows;
#pragma unroll
    for (int mh = 0; mh < 2; ++mh)
#pragma unroll
        for (int i = 0; i < 8; ++i) {
            int row = m0 + mh * 16 + i + 8 * hi;
            if (full || row < Mrows) {
#pragma unroll
                for (int j = 0; j < 4; ++j)
                    Outf[(long)row * DIMC + n0 + j * 16 + l4] = acc[mh][j][i] + bj[j];
            }
        }
}

// ---------------- depthwise 3x3x3 pool (stride 1,2,2; pad 1) + LayerNorm ---
// in: (BH, NTOK, 96) bf16. out_norm: (BH, NP, 96). out_t: (BH, 96, NPAD).
__global__ void __launch_bounds__(128)
pool_ln(const unsigned short* __restrict__ in, const float* __restrict__ kern,
        const float* __restrict__ g, const float* __restrict__ bvec,
        unsigned short* __restrict__ out_norm, unsigned short* __restrict__ out_t) {
    int w = threadIdx.x >> 5;
    long wid = (long)blockIdx.x * 4 + w;
    if (wid >= (long)BH * NP) return;
    int bh = (int)(wid / NP), tok = (int)(wid % NP);
    int lane = threadIdx.x & 31;
    float vals[3];
    if (tok == 0) {
#pragma unroll
        for (int j = 0; j < 3; ++j) {
            int c = lane + 32 * j;
            vals[j] = bf2f(in[(long)bh * NTOK * DH + c]);
        }
    } else {
        int s = tok - 1;
        int ot = s / (Ho_ * Wo_), r = s % (Ho_ * Wo_);
        int oh = r / Wo_, ow = r % Wo_;
#pragma unroll
        for (int j = 0; j < 3; ++j) {
            int c = lane + 32 * j;
            float acc = 0.f;
            for (int kt = 0; kt < 3; ++kt) {
                int it = ot - 1 + kt; if (it < 0 || it >= T0c) continue;
                for (int kh = 0; kh < 3; ++kh) {
                    int ih = oh * 2 - 1 + kh; if (ih < 0 || ih >= H0c) continue;
                    for (int kw = 0; kw < 3; ++kw) {
                        int iw = ow * 2 - 1 + kw; if (iw < 0 || iw >= W0c) continue;
                        int n = 1 + (it * H0c + ih) * W0c + iw;
                        acc += kern[c * 27 + (kt * 3 + kh) * 3 + kw] *
                               bf2f(in[((long)bh * NTOK + n) * DH + c]);
                    }
                }
            }
            vals[j] = acc;
        }
    }
    float sum = vals[0] + vals[1] + vals[2];
    float sq  = vals[0] * vals[0] + vals[1] * vals[1] + vals[2] * vals[2];
#pragma unroll
    for (int off = 16; off > 0; off >>= 1) {
        sum += __shfl_xor(sum, off, 32);
        sq  += __shfl_xor(sq,  off, 32);
    }
    float mean = sum * (1.f / 96.f);
    float var  = sq * (1.f / 96.f) - mean * mean;
    float rs   = rsqrtf(var + 1e-5f);
#pragma unroll
    for (int j = 0; j < 3; ++j) {
        int c = lane + 32 * j;
        unsigned short ub = f2bf((vals[j] - mean) * rs * g[c] + bvec[c]);
        if (out_norm) out_norm[((long)bh * NP + tok) * DH + c] = ub;
        if (out_t)    out_t[((long)bh * DH + c) * NPAD + tok] = ub;
    }
}

// ---------------- flash attention: one wave per (bh, 16-query tile) --------
// qp,kp: (BH, NP, 96) bf16 ; vt: (BH, 96, NPAD) bf16 (zero-padded keys)
// oattn: (BB, NP, 768) bf16
__global__ void __launch_bounds__(256)
attention(const unsigned short* __restrict__ qp, const unsigned short* __restrict__ kp,
          const unsigned short* __restrict__ vt, unsigned short* __restrict__ oattn) {
    __shared__ unsigned short Plds[8][16 * 32];
    const int QT = (NP + 15) / 16;  // 99
    int w = threadIdx.x >> 5;
    int wid = blockIdx.x * 8 + w;
    if (wid >= BH * QT) return;
    int bh = wid / QT, qt = wid % QT;
    int bb = bh >> 3, h = bh & 7;
    int lane = threadIdx.x & 31, hi = lane >> 4, l4 = lane & 15;
    const float scale = 0.1020620726159658f;  // 96^-0.5

    int qrow = qt * 16 + l4; if (qrow > NP - 1) qrow = NP - 1;
    const unsigned short* Qrow = qp + ((long)bh * NP + qrow) * DH;
    bf16x16 qf[3];
#pragma unroll
    for (int st = 0; st < 3; ++st) qf[st] = load_a_frag(Qrow, st * 32, hi);

    f32x8 o[6] = {};
    float mrun[8], lrun[8];
#pragma unroll
    for (int i = 0; i < 8; ++i) { mrun[i] = -1e30f; lrun[i] = 0.f; }

    for (int kb = 0; kb < NPAD; kb += 32) {
        // ---- S tile (16 queries x 32 keys), K-dim = 96 head channels ----
        f32x8 s0 = {}, s1 = {};
        int key0 = kb + l4;      if (key0 > NP - 1) key0 = NP - 1;
        int key1 = kb + 16 + l4; if (key1 > NP - 1) key1 = NP - 1;
        const unsigned short* kr0 = kp + ((long)bh * NP + key0) * DH;
        const unsigned short* kr1 = kp + ((long)bh * NP + key1) * DH;
        __builtin_prefetch(kr0 + 32 * DH, 0, 0);  // next key tile
        __builtin_prefetch(kr1 + 32 * DH, 0, 0);
#pragma unroll
        for (int st = 0; st < 3; ++st) {
            s0 = wmma_bf16(qf[st], load_b_frag(kr0 + st * 32 + hi * 16), s0);
            s1 = wmma_bf16(qf[st], load_b_frag(kr1 + st * 32 + hi * 16), s1);
        }
        // ---- online softmax ----
        bool v0 = (kb + l4)      < NP;
        bool v1 = (kb + 16 + l4) < NP;
        float a_[8], b_[8], rm[8];
#pragma unroll
        for (int i = 0; i < 8; ++i) {
            a_[i] = v0 ? s0[i] * scale : -1e30f;
            b_[i] = v1 ? s1[i] * scale : -1e30f;
            rm[i] = fmaxf(a_[i], b_[i]);
        }
#pragma unroll
        for (int off = 1; off <= 8; off <<= 1)
#pragma unroll
            for (int i = 0; i < 8; ++i) rm[i] = fmaxf(rm[i], __shfl_xor(rm[i], off, 32));
        float p0[8], p1[8], rsum[8], alpha[8];
#pragma unroll
        for (int i = 0; i < 8; ++i) {
            float mnew = fmaxf(mrun[i], rm[i]);
            alpha[i] = __expf(mrun[i] - mnew);
            p0[i] = __expf(a_[i] - mnew);
            p1[i] = __expf(b_[i] - mnew);
            rsum[i] = p0[i] + p1[i];
            mrun[i] = mnew;
        }
#pragma unroll
        for (int off = 1; off <= 8; off <<= 1)
#pragma unroll
            for (int i = 0; i < 8; ++i) rsum[i] += __shfl_xor(rsum[i], off, 32);
#pragma unroll
        for (int i = 0; i < 8; ++i) lrun[i] = lrun[i] * alpha[i] + rsum[i];
#pragma unroll
        for (int j = 0; j < 6; ++j)
#pragma unroll
            for (int i = 0; i < 8; ++i) o[j][i] *= alpha[i];
        // ---- redistribute P (C-layout -> A-layout) via LDS ----
#pragma unroll
        for (int i = 0; i < 8; ++i) {
            int row = i + 8 * hi;
            Plds[w][row * 32 + l4]      = f2bf(p0[i]);
            Plds[w][row * 32 + 16 + l4] = f2bf(p1[i]);
        }
        asm volatile("s_wait_dscnt 0x0" ::: "memory");
        bf16x16 pf;
#pragma unroll
        for (int i = 0; i < 8; ++i) {
            pf[i]     = us2bf(Plds[w][l4 * 32 + hi * 16 + i]);
            pf[8 + i] = us2bf(Plds[w][l4 * 32 + hi * 16 + 8 + i]);
        }
        // ---- O += P @ V  (B from d-major, key-padded vt: contiguous loads) ----
#pragma unroll
        for (int j = 0; j < 6; ++j) {
            const unsigned short* vp =
                vt + ((long)bh * DH + j * 16 + l4) * NPAD + kb + hi * 16;
            o[j] = wmma_bf16(pf, load_b_frag(vp), o[j]);
        }
    }
    // ---- finalize: divide by l, scatter to (B, NP, 768) ----
    float inv[8];
#pragma unroll
    for (int i = 0; i < 8; ++i) inv[i] = lrun[i] > 0.f ? 1.f / lrun[i] : 0.f;
#pragma unroll
    for (int j = 0; j < 6; ++j)
#pragma unroll
        for (int i = 0; i < 8; ++i) {
            int row = qt * 16 + i + 8 * hi;
            if (row < NP) {
                int d = j * 16 + l4;
                oattn[((long)bb * NP + row) * DIMC + h * DH + d] =
                    f2bf(o[j][i] * inv[i]);
            }
        }
}

// ---------------- host launcher ----------------
extern "C" void kernel_launch(void* const* d_in, const int* in_sizes, int n_in,
                              void* d_out, int out_size, void* d_ws, size_t ws_size,
                              hipStream_t stream) {
    const float* x     = (const float*)d_in[0];
    const float* Wq    = (const float*)d_in[1];
    const float* Wk    = (const float*)d_in[2];
    const float* Wv    = (const float*)d_in[3];
    const float* Wp    = (const float*)d_in[4];
    const float* bproj = (const float*)d_in[5];
    const float* kq    = (const float*)d_in[6];
    const float* kk    = (const float*)d_in[7];
    const float* kv    = (const float*)d_in[8];
    const float* gq = (const float*)d_in[9],  *bq = (const float*)d_in[10];
    const float* gk = (const float*)d_in[11], *bk = (const float*)d_in[12];
    const float* gv = (const float*)d_in[13], *bv = (const float*)d_in[14];

    char* ws = (char*)d_ws;
    size_t off = 0;
    auto take = [&](size_t nelem) -> unsigned short* {
        unsigned short* p = (unsigned short*)(ws + off);
        off = (off + nelem * 2 + 255) & ~(size_t)255;
        return p;
    };
    const size_t NX = (size_t)MQKV * DIMC;      // x / q / k / v  elements
    const size_t NW = (size_t)DIMC * DIMC;      // weight elements
    const size_t NPOOL = (size_t)BH * NP * DH;  // pooled q/k
    const size_t NVT = (size_t)BH * DH * NPAD;  // v transposed+padded
    const size_t NO = (size_t)MPROJ * DIMC;     // attention output

    unsigned short* xb  = take(NX);
    unsigned short* wqb = take(NW);
    unsigned short* wkb = take(NW);
    unsigned short* wvb = take(NW);
    unsigned short* wpb = take(NW);
    unsigned short* qb  = take(NX);
    unsigned short* kbf = take(NX);
    unsigned short* vbf = take(NX);
    unsigned short* qpb = take(NPOOL);
    unsigned short* kpb = take(NPOOL);
    unsigned short* vtb = take(NVT);
    unsigned short* oat = take(NO);

    // 1) f32 -> bf16 conversions (4-wide)
    long nx4 = (long)(NX / 4), nw4 = (long)(NW / 4), nvt4 = (long)(NVT / 4);
    cvt_f32_bf16_v4<<<(int)((nx4 + 255) / 256), 256, 0, stream>>>(x, xb, nx4);
    cvt_f32_bf16_v4<<<(int)((nw4 + 255) / 256), 256, 0, stream>>>(Wq, wqb, nw4);
    cvt_f32_bf16_v4<<<(int)((nw4 + 255) / 256), 256, 0, stream>>>(Wk, wkb, nw4);
    cvt_f32_bf16_v4<<<(int)((nw4 + 255) / 256), 256, 0, stream>>>(Wv, wvb, nw4);
    cvt_f32_bf16_v4<<<(int)((nw4 + 255) / 256), 256, 0, stream>>>(Wp, wpb, nw4);
    zero_u16_v4<<<(int)((nvt4 + 255) / 256), 256, 0, stream>>>(vtb, nvt4);

    // 2) QKV projections (WMMA bf16, 32x64 per wave)
    dim3 gq_grid((((MQKV + 31) / 32) + 7) / 8, DIMC / 64);
    gemm_qkv<<<gq_grid, 256, 0, stream>>>(xb, wqb, qb,  MQKV);
    gemm_qkv<<<gq_grid, 256, 0, stream>>>(xb, wkb, kbf, MQKV);
    gemm_qkv<<<gq_grid, 256, 0, stream>>>(xb, wvb, vbf, MQKV);

    // 3) attention pooling + layernorm
    int pool_blocks = (int)(((long)BH * NP + 3) / 4);
    pool_ln<<<pool_blocks, 128, 0, stream>>>(qb,  kq, gq, bq, qpb, nullptr);
    pool_ln<<<pool_blocks, 128, 0, stream>>>(kbf, kk, gk, bk, kpb, nullptr);
    pool_ln<<<pool_blocks, 128, 0, stream>>>(vbf, kv, gv, bv, nullptr, vtb);

    // 4) flash attention (WMMA bf16)
    int qtiles = (NP + 15) / 16;               // 99
    int att_blocks = (BH * qtiles + 7) / 8;    // 396
    attention<<<att_blocks, 256, 0, stream>>>(qpb, kpb, vtb, oat);

    // 5) output projection + bias (WMMA bf16, f32 out)
    dim3 gp_grid((((MPROJ + 31) / 32) + 7) / 8, DIMC / 64);
    gemm_proj<<<gp_grid, 256, 0, stream>>>(oat, wpb, bproj, (float*)d_out, MPROJ);
}
